// EdgeModel_68848325755267
// MI455X (gfx1250) — matrix operand order repaired and checked by reading, not compiled, not run on hardware
//
#include <hip/hip_runtime.h>
#include <hip/hip_bf16.h>
#include <math.h>

typedef __attribute__((ext_vector_type(16))) _Float16 v16h;
typedef __attribute__((ext_vector_type(8)))  _Float16 v8h;
typedef __attribute__((ext_vector_type(8)))  float    v8f;

#define FDIM   128
#define NVERT  512
#define CDIM   10976
#define EPS_BN 1e-5f

__device__ __forceinline__ float elu1(float x){ return x > 0.f ? x : (__expf(x) - 1.f); }

__device__ __forceinline__ void zero4(v8f c[4]){
#pragma unroll
  for (int t = 0; t < 4; ++t)
#pragma unroll
    for (int e = 0; e < 8; ++e) c[t][e] = 0.f;
}

// ---- WMMA fragment loaders (v_wmma_f32_16x16x32_f16 layouts per CDNA5 ISA 7.12.2) ----
// A: 16x32 f16 row-major tile (stride ld halves). lane: m=lane&15, g=lane>>4.
// element e<8  -> K = g*8 + e ; e>=8 -> K = 16 + g*8 + (e-8)
__device__ __forceinline__ v16h frag_a(const _Float16* base, int ld, int lane){
  int m = lane & 15, g = lane >> 4;
  const _Float16* p = base + m * ld + g * 8;
  v8h lo = *(const v8h*)(p);
  v8h hi = *(const v8h*)(p + 16);
  v16h r;
#pragma unroll
  for (int e = 0; e < 8; ++e){ r[e] = lo[e]; r[e + 8] = hi[e]; }
  return r;
}
// B: 32x16 f16. Weights stored transposed Wt[N][K]. lane: n=n0+(lane&15), g=lane>>4.
// element e<8 -> K = kb + g*8 + e ; e>=8 -> K = kb + 16 + g*8 + (e-8)
__device__ __forceinline__ v16h frag_b(const _Float16* Wt, int K, int kb, int n0, int lane){
  int n = n0 + (lane & 15), g = lane >> 4;
  const _Float16* p = Wt + (size_t)n * K + kb + g * 8;
  v8h lo = *(const v8h*)(p);
  v8h hi = *(const v8h*)(p + 16);
  v16h r;
#pragma unroll
  for (int e = 0; e < 8; ++e){ r[e] = lo[e]; r[e + 8] = hi[e]; }
  return r;
}

// 8-wave workgroup GEMM: OUT(64 x 128) += A(64 x K, f16 LDS row-major) * W (K x 128) given as Wt[128][K]
// wave w: m-subtile mt=w&3, n-tiles { (w>>2) + 2t : t<4 }. C f32 layout: lane ln=lane&15 -> n, vgpr v + (lane>>4)*8 -> m.
__device__ __forceinline__ void wg_gemm64(const _Float16* aLds, int K, const _Float16* Wt, v8f c[4]){
  int lane = threadIdx.x & 31, wave = threadIdx.x >> 5;
  int mt = wave & 3, ntg = wave >> 2;
  const _Float16* aBase = aLds + mt * 16 * K;
  for (int kb = 0; kb < K; kb += 32){
    v16h a = frag_a(aBase + kb, K, lane);
#pragma unroll
    for (int t = 0; t < 4; ++t){
      v16h b = frag_b(Wt, K, kb, (ntg + 2 * t) * 16, lane);
      c[t] = __builtin_amdgcn_wmma_f32_16x16x32_f16(false, a, false, b, (short)0, c[t], false, false);
    }
  }
}

__device__ __forceinline__ void ep_bias_elu_f16(v8f c[4], const float* bias, _Float16* outLds){
  int lane = threadIdx.x & 31, wave = threadIdx.x >> 5;
  int mt = wave & 3, ntg = wave >> 2, g = lane >> 4, ln = lane & 15;
#pragma unroll
  for (int t = 0; t < 4; ++t){
    int n = (ntg + 2 * t) * 16 + ln;
    float b = bias[n];
#pragma unroll
    for (int v = 0; v < 8; ++v)
      outLds[(mt * 16 + g * 8 + v) * FDIM + n] = (_Float16)elu1(c[t][v] + b);
  }
}

__device__ __forceinline__ void ep_bias_elu_bn(v8f c[4], const float* bias, const float* bg,
                                               const float* bbe, const float* bm, const float* bv){
  int lane = threadIdx.x & 31, wave = threadIdx.x >> 5;
  int ntg = wave >> 2, ln = lane & 15;
#pragma unroll
  for (int t = 0; t < 4; ++t){
    int n = (ntg + 2 * t) * 16 + ln;
    float b  = bias[n];
    float sc = bg[n] * rsqrtf(bv[n] + EPS_BN);
    float sh = bbe[n] - bm[n] * sc;
#pragma unroll
    for (int v = 0; v < 8; ++v)
      c[t][v] = elu1(c[t][v] + b) * sc + sh;
  }
}

__device__ __forceinline__ void ep_elu_bn_store_g(v8f c[4], const float* bias, const float* bg,
                                                  const float* bbe, const float* bm, const float* bv,
                                                  float* out, int ldo, int r0, int col0){
  ep_bias_elu_bn(c, bias, bg, bbe, bm, bv);
  int lane = threadIdx.x & 31, wave = threadIdx.x >> 5;
  int mt = wave & 3, ntg = wave >> 2, g = lane >> 4, ln = lane & 15;
#pragma unroll
  for (int t = 0; t < 4; ++t){
    int n = (ntg + 2 * t) * 16 + ln;
#pragma unroll
    for (int v = 0; v < 8; ++v)
      out[(size_t)(r0 + mt * 16 + g * 8 + v) * ldo + col0 + n] = c[t][v];
  }
}

// ---------------- weight convert: f32 [K][N] -> f16 [N][K] ----------------
__global__ void k_wcvt(const float* __restrict__ src, _Float16* __restrict__ dst, int K, int N){
  int e = blockIdx.x * 256 + threadIdx.x;
  if (e < K * N){
    int k = e / N, n = e - k * N;
    dst[(size_t)n * K + k] = (_Float16)src[e];
  }
}

// ---------------- vertex features: vertf[512][256] = [vf | ff] ----------------
__global__ void __launch_bounds__(256) k_vert(
    const float* __restrict__ verts, const float* __restrict__ f2,
    const float* __restrict__ mvW1, const float* __restrict__ mvb1,
    const _Float16* __restrict__ mvW2t, const float* __restrict__ mvb2,
    const float* mvg, const float* mvbe, const float* mvm, const float* mvv,
    const _Float16* __restrict__ mfW1t, const float* __restrict__ mfb1,
    const _Float16* __restrict__ mfW2t, const float* __restrict__ mfb2,
    const float* mfg, const float* mfbe, const float* mfm, const float* mfv,
    float* __restrict__ vertf)
{
  __shared__ __align__(32) _Float16 h1h[64 * FDIM];
  __shared__ __align__(32) _Float16 Achunk[64 * 32];
  __shared__ float vLds[64 * 3];
  __shared__ unsigned rowOff[64];
  int t  = threadIdx.x;
  int r0 = blockIdx.x * 64;

  if (t < 192) vLds[t] = verts[r0 * 3 + t];
  if (t < 64){
    float x = verts[(r0 + t) * 3 + 0];
    float y = verts[(r0 + t) * 3 + 1];
    float z = verts[(r0 + t) * 3 + 2];
    int ix = (int)((x + 1.f) * 6.f); ix = ix < 0 ? 0 : (ix > 12 ? 12 : ix);
    int iy = (int)((y + 1.f) * 6.f); iy = iy < 0 ? 0 : (iy > 12 ? 12 : iy);
    int iz = (int)((z + 1.f) * 6.f); iz = iz < 0 ? 0 : (iz > 12 ? 12 : iz);
    rowOff[t] = (unsigned)(((ix * 13 + iy) * 13 + iz)) * CDIM;
  }
  __syncthreads();

  // ---- vf: layer1 (K=3, VALU) ----
  for (int q = 0; q < 32; ++q){
    int e = t + 256 * q; int m = e >> 7, n = e & 127;
    float s = vLds[m * 3 + 0] * mvW1[n] + vLds[m * 3 + 1] * mvW1[FDIM + n] +
              vLds[m * 3 + 2] * mvW1[2 * FDIM + n] + mvb1[n];
    h1h[m * FDIM + n] = (_Float16)elu1(s);
  }
  __syncthreads();
  {
    v8f c[4]; zero4(c);
    wg_gemm64(h1h, FDIM, mvW2t, c);
    ep_elu_bn_store_g(c, mvb2, mvg, mvbe, mvm, mvv, vertf, 256, r0, 0);
  }
  __syncthreads();

  // ---- ff: layer1 (K=10976 streamed through LDS in K-chunks of 32) ----
  v8f cf[4]; zero4(cf);
  int lane = t & 31, wave = t >> 5, mt = wave & 3, ntg = wave >> 2;
  for (int kb = 0; kb < CDIM; kb += 32){
#pragma unroll
    for (int q = 0; q < 8; ++q){
      int e = t + 256 * q; int m = e >> 5, kk = e & 31;
      Achunk[m * 32 + kk] = (_Float16)f2[(size_t)rowOff[m] + kb + kk];
    }
    __syncthreads();
    v16h a = frag_a(Achunk + mt * 16 * 32, 32, lane);
#pragma unroll
    for (int tt = 0; tt < 4; ++tt){
      v16h b = frag_b(mfW1t, CDIM, kb, (ntg + 2 * tt) * 16, lane);
      cf[tt] = __builtin_amdgcn_wmma_f32_16x16x32_f16(false, a, false, b, (short)0, cf[tt], false, false);
    }
    __syncthreads();
  }
  ep_bias_elu_f16(cf, mfb1, h1h);
  __syncthreads();
  {
    v8f c2[4]; zero4(c2);
    wg_gemm64(h1h, FDIM, mfW2t, c2);
    ep_elu_bn_store_g(c2, mfb2, mfg, mfbe, mfm, mfv, vertf, 256, r0, 128);
  }
}

// ---------------- pass1: s1 tiles + deterministic weighted node partials ----------------
__global__ void __launch_bounds__(256) k_pass1(
    const float* __restrict__ vertf, const float* __restrict__ adj,
    const _Float16* __restrict__ s1W1t, const float* __restrict__ s1b1,
    const _Float16* __restrict__ s1W2t, const float* __restrict__ s1b2,
    const float* s1g, const float* s1be, const float* s1m, const float* s1v,
    float* __restrict__ partials)
{
  __shared__ __align__(32) _Float16 e1h[64 * 256];
  __shared__ __align__(32) _Float16 h1h[64 * FDIM];
  __shared__ float viLds[256];
  __shared__ float adjLds[64];
  __shared__ float nodePart[4][FDIM];
  int t = threadIdx.x;
  int i = blockIdx.y, jb = blockIdx.x, j0 = jb * 64;

  viLds[t] = vertf[(size_t)i * 256 + t];
  if (t < 64) adjLds[t] = adj[(size_t)i * NVERT + j0 + t];
  __syncthreads();
  for (int m = 0; m < 64; ++m)
    e1h[m * 256 + t] = (_Float16)(viLds[t] + vertf[(size_t)(j0 + m) * 256 + t]);
  __syncthreads();

  v8f c1[4]; zero4(c1);
  wg_gemm64(e1h, 256, s1W1t, c1);
  ep_bias_elu_f16(c1, s1b1, h1h);
  __syncthreads();

  v8f c2[4]; zero4(c2);
  wg_gemm64(h1h, FDIM, s1W2t, c2);
  ep_bias_elu_bn(c2, s1b2, s1g, s1be, s1m, s1v);

  // weighted reduction over the 64 j's of this block (fixed-order, unique writers)
  int lane = t & 31, wave = t >> 5, mt = wave & 3, ntg = wave >> 2, g = lane >> 4, ln = lane & 15;
#pragma unroll
  for (int t4 = 0; t4 < 4; ++t4){
    float p = 0.f;
#pragma unroll
    for (int v = 0; v < 8; ++v)
      p += c2[t4][v] * adjLds[mt * 16 + g * 8 + v];
    p += __shfl_xor(p, 16, 32);
    if (lane < 16) nodePart[mt][(ntg + 2 * t4) * 16 + ln] = p;
  }
  __syncthreads();
  if (t < FDIM){
    float s = nodePart[0][t] + nodePart[1][t] + nodePart[2][t] + nodePart[3][t];
    partials[((size_t)i * 8 + jb) * FDIM + t] = s;
  }
}

__global__ void k_nred(const float* __restrict__ partials, float* __restrict__ nodes){
  int i = blockIdx.x, t = threadIdx.x;
  float s = 0.f;
  for (int jb = 0; jb < 8; ++jb) s += partials[((size_t)i * 8 + jb) * FDIM + t];
  nodes[(size_t)i * FDIM + t] = s;
}

// ---------------- n2 = block(nodes; s2) ----------------
__global__ void __launch_bounds__(256) k_n2(
    const float* __restrict__ nodes,
    const _Float16* __restrict__ s2W1t, const float* __restrict__ s2b1,
    const _Float16* __restrict__ s2W2t, const float* __restrict__ s2b2,
    const float* s2g, const float* s2be, const float* s2m, const float* s2v,
    float* __restrict__ n2)
{
  __shared__ __align__(32) _Float16 inh[64 * FDIM];
  __shared__ __align__(32) _Float16 h1h[64 * FDIM];
  int t = threadIdx.x, r0 = blockIdx.x * 64;
  for (int q = 0; q < 32; ++q){
    int e = t + 256 * q; int m = e >> 7, n = e & 127;
    inh[m * FDIM + n] = (_Float16)nodes[(size_t)(r0 + m) * FDIM + n];
  }
  __syncthreads();
  v8f c1[4]; zero4(c1);
  wg_gemm64(inh, FDIM, s2W1t, c1);
  ep_bias_elu_f16(c1, s2b1, h1h);
  __syncthreads();
  v8f c2[4]; zero4(c2);
  wg_gemm64(h1h, FDIM, s2W2t, c2);
  ep_elu_bn_store_g(c2, s2b2, s2g, s2be, s2m, s2v, n2, FDIM, r0, 0);
}

// ---------------- pass3: recompute s1, cat with s2, s3 block, 128->2 head ----------------
__global__ void __launch_bounds__(256) k_pass3(
    const float* __restrict__ vertf, const float* __restrict__ n2,
    const _Float16* __restrict__ s1W1t, const float* __restrict__ s1b1,
    const _Float16* __restrict__ s1W2t, const float* __restrict__ s1b2,
    const float* s1g, const float* s1be, const float* s1m, const float* s1v,
    const _Float16* __restrict__ s3W1t, const float* __restrict__ s3b1,
    const _Float16* __restrict__ s3W2t, const float* __restrict__ s3b2,
    const float* s3g, const float* s3be, const float* s3m, const float* s3v,
    const float* __restrict__ s3W3, const float* __restrict__ s3b3,
    float* __restrict__ out)
{
  // union buffer: e1h(f16 64x256) -> cath(f16 64x256) -> hF(f32 64x128), all 32 KB, lifetimes separated by barriers
  __shared__ __align__(32) unsigned char uraw[64 * 256 * sizeof(_Float16)];
  __shared__ __align__(32) _Float16 h1h[64 * FDIM];
  __shared__ float viLds[256];
  __shared__ float n2i[FDIM];
  _Float16* e1h = (_Float16*)uraw;
  int t = threadIdx.x;
  int i = blockIdx.y, jb = blockIdx.x, j0 = jb * 64;

  viLds[t] = vertf[(size_t)i * 256 + t];
  if (t < FDIM) n2i[t] = n2[(size_t)i * FDIM + t];
  __syncthreads();
  for (int m = 0; m < 64; ++m)
    e1h[m * 256 + t] = (_Float16)(viLds[t] + vertf[(size_t)(j0 + m) * 256 + t]);
  __syncthreads();

  // s1 block
  v8f c1[4]; zero4(c1);
  wg_gemm64(e1h, 256, s1W1t, c1);
  ep_bias_elu_f16(c1, s1b1, h1h);
  __syncthreads();                       // e1h dead after this point
  v8f c2[4]; zero4(c2);
  wg_gemm64(h1h, FDIM, s1W2t, c2);
  ep_bias_elu_bn(c2, s1b2, s1g, s1be, s1m, s1v);

  _Float16* cath = (_Float16*)uraw;      // reuse: cat = [s1 | n2[i]+n2[j]]
  {
    int lane = t & 31, wave = t >> 5, mt = wave & 3, ntg = wave >> 2, g = lane >> 4, ln = lane & 15;
#pragma unroll
    for (int t4 = 0; t4 < 4; ++t4){
      int n = (ntg + 2 * t4) * 16 + ln;
#pragma unroll
      for (int v = 0; v < 8; ++v)
        cath[(mt * 16 + g * 8 + v) * 256 + n] = (_Float16)c2[t4][v];
    }
  }
  for (int q = 0; q < 32; ++q){
    int e = t + 256 * q; int m = e >> 7, n = e & 127;
    cath[m * 256 + 128 + n] = (_Float16)(n2i[n] + n2[(size_t)(j0 + m) * FDIM + n]);
  }
  __syncthreads();

  // s3 block
  v8f c3[4]; zero4(c3);
  wg_gemm64(cath, 256, s3W1t, c3);
  ep_bias_elu_f16(c3, s3b1, h1h);
  __syncthreads();                       // cath dead after this point
  v8f c4[4]; zero4(c4);
  wg_gemm64(h1h, FDIM, s3W2t, c4);
  ep_bias_elu_bn(c4, s3b2, s3g, s3be, s3m, s3v);

  float* hF = (float*)uraw;              // reuse as f32 64x128
  {
    int lane = t & 31, wave = t >> 5, mt = wave & 3, ntg = wave >> 2, g = lane >> 4, ln = lane & 15;
#pragma unroll
    for (int t4 = 0; t4 < 4; ++t4){
      int n = (ntg + 2 * t4) * 16 + ln;
#pragma unroll
      for (int v = 0; v < 8; ++v)
        hF[(mt * 16 + g * 8 + v) * FDIM + n] = c4[t4][v];
    }
  }
  __syncthreads();

  // final head: 128 -> 2
  if (t < 128){
    int m = t >> 1, cc = t & 1;
    float s = s3b3[cc];
    for (int n = 0; n < FDIM; ++n) s += hF[m * FDIM + n] * s3W3[n * 2 + cc];
    out[(((size_t)i * NVERT) + j0 + m) * 2 + cc] = s;
  }
}

// ======================================================================================
extern "C" void kernel_launch(void* const* d_in, const int* in_sizes, int n_in,
                              void* d_out, int out_size, void* d_ws, size_t ws_size,
                              hipStream_t stream)
{
  (void)in_sizes; (void)n_in; (void)out_size; (void)ws_size;
  const float* verts = (const float*)d_in[0];
  const float* f2    = (const float*)d_in[1];
  const float* adj   = (const float*)d_in[4];

  // dict order: per block p in {mv,mf,s1,s2,s3}: W1,b1,W2,b2,g,be,m,v
  const float* mvW1 = (const float*)d_in[5];  const float* mvb1 = (const float*)d_in[6];
  const float* mvW2 = (const float*)d_in[7];  const float* mvb2 = (const float*)d_in[8];
  const float* mvg  = (const float*)d_in[9];  const float* mvbe = (const float*)d_in[10];
  const float* mvm  = (const float*)d_in[11]; const float* mvv  = (const float*)d_in[12];
  const float* mfW1 = (const float*)d_in[13]; const float* mfb1 = (const float*)d_in[14];
  const float* mfW2 = (const float*)d_in[15]; const float* mfb2 = (const float*)d_in[16];
  const float* mfg  = (const float*)d_in[17]; const float* mfbe = (const float*)d_in[18];
  const float* mfm  = (const float*)d_in[19]; const float* mfv  = (const float*)d_in[20];
  const float* s1W1 = (const float*)d_in[21]; const float* s1b1 = (const float*)d_in[22];
  const float* s1W2 = (const float*)d_in[23]; const float* s1b2 = (const float*)d_in[24];
  const float* s1g  = (const float*)d_in[25]; const float* s1be = (const float*)d_in[26];
  const float* s1m  = (const float*)d_in[27]; const float* s1v  = (const float*)d_in[28];
  const float* s2W1 = (const float*)d_in[29]; const float* s2b1 = (const float*)d_in[30];
  const float* s2W2 = (const float*)d_in[31]; const float* s2b2 = (const float*)d_in[32];
  const float* s2g  = (const float*)d_in[33]; const float* s2be = (const float*)d_in[34];
  const float* s2m  = (const float*)d_in[35]; const float* s2v  = (const float*)d_in[36];
  const float* s3W1 = (const float*)d_in[37]; const float* s3b1 = (const float*)d_in[38];
  const float* s3W2 = (const float*)d_in[39]; const float* s3b2 = (const float*)d_in[40];
  const float* s3g  = (const float*)d_in[41]; const float* s3be = (const float*)d_in[42];
  const float* s3m  = (const float*)d_in[43]; const float* s3v  = (const float*)d_in[44];
  const float* s3W3 = (const float*)d_in[45]; const float* s3b3 = (const float*)d_in[46];

  unsigned char* ws = (unsigned char*)d_ws;
  size_t off = 0;
  auto alloc = [&](size_t bytes) -> void* {
    void* p = ws + off;
    off = (off + bytes + 255) & ~(size_t)255;
    return p;
  };
  float*     vertf    = (float*)alloc((size_t)NVERT * 256 * 4);
  _Float16*  mvW2t    = (_Float16*)alloc((size_t)128 * 128 * 2);
  _Float16*  mfW1t    = (_Float16*)alloc((size_t)CDIM * 128 * 2);
  _Float16*  mfW2t    = (_Float16*)alloc((size_t)128 * 128 * 2);
  _Float16*  s1W1t    = (_Float16*)alloc((size_t)256 * 128 * 2);
  _Float16*  s1W2t    = (_Float16*)alloc((size_t)128 * 128 * 2);
  _Float16*  s2W1t    = (_Float16*)alloc((size_t)128 * 128 * 2);
  _Float16*  s2W2t    = (_Float16*)alloc((size_t)128 * 128 * 2);
  _Float16*  s3W1t    = (_Float16*)alloc((size_t)256 * 128 * 2);
  _Float16*  s3W2t    = (_Float16*)alloc((size_t)128 * 128 * 2);
  float*     partials = (float*)alloc((size_t)NVERT * 8 * FDIM * 4);
  float*     nodes    = (float*)alloc((size_t)NVERT * FDIM * 4);
  float*     n2buf    = (float*)alloc((size_t)NVERT * FDIM * 4);

  auto cvt = [&](const float* src, _Float16* dst, int K, int Nn){
    int tot = K * Nn;
    k_wcvt<<<(tot + 255) / 256, 256, 0, stream>>>(src, dst, K, Nn);
  };
  cvt(mvW2, mvW2t, 128, 128);
  cvt(mfW1, mfW1t, CDIM, 128);
  cvt(mfW2, mfW2t, 128, 128);
  cvt(s1W1, s1W1t, 256, 128);
  cvt(s1W2, s1W2t, 128, 128);
  cvt(s2W1, s2W1t, 128, 128);
  cvt(s2W2, s2W2t, 128, 128);
  cvt(s3W1, s3W1t, 256, 128);
  cvt(s3W2, s3W2t, 128, 128);

  k_vert<<<NVERT / 64, 256, 0, stream>>>(verts, f2,
      mvW1, mvb1, mvW2t, mvb2, mvg, mvbe, mvm, mvv,
      mfW1t, mfb1, mfW2t, mfb2, mfg, mfbe, mfm, mfv, vertf);

  k_pass1<<<dim3(8, NVERT), 256, 0, stream>>>(vertf, adj,
      s1W1t, s1b1, s1W2t, s1b2, s1g, s1be, s1m, s1v, partials);

  k_nred<<<NVERT, FDIM, 0, stream>>>(partials, nodes);

  k_n2<<<NVERT / 64, 256, 0, stream>>>(nodes,
      s2W1t, s2b1, s2W2t, s2b2, s2g, s2be, s2m, s2v, n2buf);

  k_pass3<<<dim3(8, NVERT), 256, 0, stream>>>(vertf, n2buf,
      s1W1t, s1b1, s1W2t, s1b2, s1g, s1be, s1m, s1v,
      s3W1t, s3b1, s3W2t, s3b2, s3g, s3be, s3m, s3v,
      s3W3, s3b3, (float*)d_out);
}
